// AttentionDecoder_22428319220783
// MI455X (gfx1250) — matrix operand order repaired and checked by reading, not compiled, not run on hardware
//
#include <hip/hip_runtime.h>
#include <cstddef>

// Problem sizes (fixed by the reference)
#define N_ 32
#define T_ 2048
#define H_ 512
#define C_ 512
#define D_ 256
#define E_ 512
#define V_ 50257

typedef __attribute__((ext_vector_type(16))) _Float16 v16h;
typedef __attribute__((ext_vector_type(8)))  float    v8f;

#if __has_builtin(__builtin_amdgcn_tanhf)
__device__ __forceinline__ float fast_tanhf(float x) { return __builtin_amdgcn_tanhf(x); }
#else
__device__ __forceinline__ float fast_tanhf(float x) { return tanhf(x); }
#endif

__device__ __forceinline__ float sigmoidf_(float x) { return 1.0f / (1.0f + expf(-x)); }

// ---------------------------------------------------------------------------
// r_t[n,d] = rb[d] + sum_k h0[n,k] * rW[d,k]      (32x256, tiny)
__global__ void k_rt(const float* __restrict__ h0, const float* __restrict__ rW,
                     const float* __restrict__ rb, float* __restrict__ rt) {
    int n = blockIdx.x;
    int d = threadIdx.x; // 256 threads = D_
    __shared__ float hs[H_];
    for (int k = threadIdx.x; k < H_; k += blockDim.x) hs[k] = h0[n * H_ + k];
    __syncthreads();
    const float* w = rW + (size_t)d * H_;
    float s = rb[d];
    #pragma unroll 4
    for (int k = 0; k < H_; ++k) s += hs[k] * w[k];
    rt[n * D_ + d] = s;
}

// ---------------------------------------------------------------------------
// u[n,t] = (sum_d tanh(wf[n,t,d] + rt[n,d]) * uW[d] + ub) * mask[n,t]
// One wave32 per (n,t); lane covers 8 of D=256 via two coalesced float4 loads.
__global__ void k_scores(const float* __restrict__ wf, const float* __restrict__ rt,
                         const float* __restrict__ uW, const float* __restrict__ ub,
                         const float* __restrict__ mask, float* __restrict__ u) {
    int wid  = blockIdx.x * 8 + (threadIdx.x >> 5);   // global wave id = n*T + t
    int lane = threadIdx.x & 31;
    int n = wid >> 11;            // T_ = 2048
    int t = wid & (T_ - 1);
    const float* p = wf + ((size_t)(n * T_ + t)) * D_;
    const float* r = rt + n * D_;
    float s = 0.0f;
    #pragma unroll
    for (int c = 0; c < 2; ++c) {
        int d = c * 128 + lane * 4;
        float4 w4 = *(const float4*)(p  + d);
        float4 r4 = *(const float4*)(r  + d);
        float4 q4 = *(const float4*)(uW + d);
        s += fast_tanhf(w4.x + r4.x) * q4.x + fast_tanhf(w4.y + r4.y) * q4.y
           + fast_tanhf(w4.z + r4.z) * q4.z + fast_tanhf(w4.w + r4.w) * q4.w;
    }
    #pragma unroll
    for (int off = 16; off > 0; off >>= 1) s += __shfl_xor(s, off, 32);
    if (lane == 0) u[n * T_ + t] = (s + ub[0]) * mask[n * T_ + t];
}

// ---------------------------------------------------------------------------
// in-place softmax over T per row n
__global__ void k_softmax(float* __restrict__ u) {
    int n = blockIdx.x;
    float* row = u + n * T_;
    __shared__ float red[256];
    float m = -1e30f;
    for (int t = threadIdx.x; t < T_; t += 256) m = fmaxf(m, row[t]);
    red[threadIdx.x] = m; __syncthreads();
    for (int s = 128; s > 0; s >>= 1) {
        if (threadIdx.x < s) red[threadIdx.x] = fmaxf(red[threadIdx.x], red[threadIdx.x + s]);
        __syncthreads();
    }
    m = red[0]; __syncthreads();
    float sum = 0.0f;
    for (int t = threadIdx.x; t < T_; t += 256) { float e = expf(row[t] - m); row[t] = e; sum += e; }
    red[threadIdx.x] = sum; __syncthreads();
    for (int s = 128; s > 0; s >>= 1) {
        if (threadIdx.x < s) red[threadIdx.x] += red[threadIdx.x + s];
        __syncthreads();
    }
    float inv = 1.0f / red[0];
    for (int t = threadIdx.x; t < T_; t += 256) row[t] *= inv;
}

// ---------------------------------------------------------------------------
// part[n,chunk,c] = sum_{t in chunk} a[n,t] * enc[n,t,c]   (deterministic)
__global__ void k_context(const float* __restrict__ a, const float* __restrict__ enc,
                          float* __restrict__ part) {
    int n = blockIdx.x >> 3, chunk = blockIdx.x & 7;
    int c = threadIdx.x;                       // 256 threads, each covers c and c+256
    int t0 = chunk * (T_ / 8);
    float a0 = 0.0f, a1 = 0.0f;
    for (int t = t0; t < t0 + T_ / 8; ++t) {
        float av = a[n * T_ + t];              // block-uniform -> scalar load
        const float* e = enc + ((size_t)(n * T_ + t)) * C_;
        __builtin_prefetch(e + 8 * C_, 0, 1);  // global_prefetch_b8 a few rows ahead
        a0 += av * e[c];
        a1 += av * e[c + 256];
    }
    part[(size_t)(n * 8 + chunk) * C_ + c]       = a0;
    part[(size_t)(n * 8 + chunk) * C_ + c + 256] = a1;
}

// ctx[n,c] = sum_chunk part[n,chunk,c]
__global__ void k_ctx_reduce(const float* __restrict__ part, float* __restrict__ ctx) {
    int i = blockIdx.x * blockDim.x + threadIdx.x;    // N*C
    int n = i >> 9, c = i & (C_ - 1);
    float s = 0.0f;
    #pragma unroll
    for (int k = 0; k < 8; ++k) s += part[(size_t)(n * 8 + k) * C_ + c];
    ctx[i] = s;
}

// ---------------------------------------------------------------------------
// x[n,:] = relu([emb_table[ids[n]], ctx[n]])   -> [32, E+C]
__global__ void k_build_x(const int* __restrict__ ids, const float* __restrict__ emb,
                          const float* __restrict__ ctx, float* __restrict__ x) {
    int n = blockIdx.x;
    const float* er = emb + (size_t)ids[n] * E_;
    for (int j = threadIdx.x; j < E_ + C_; j += blockDim.x) {
        float v = (j < E_) ? er[j] : ctx[n * C_ + (j - E_)];
        x[n * (E_ + C_) + j] = fmaxf(v, 0.0f);
    }
}

// ---------------------------------------------------------------------------
// WMMA fragment helpers (ISA 7.12.2 layouts, f32 -> f16 convert in-register)
__device__ __forceinline__ v16h cvt16(float4 x0, float4 x1, float4 x2, float4 x3) {
    v16h a;
    a[0] = (_Float16)x0.x; a[1]  = (_Float16)x0.y; a[2]  = (_Float16)x0.z; a[3]  = (_Float16)x0.w;
    a[4] = (_Float16)x1.x; a[5]  = (_Float16)x1.y; a[6]  = (_Float16)x1.z; a[7]  = (_Float16)x1.w;
    a[8] = (_Float16)x2.x; a[9]  = (_Float16)x2.y; a[10] = (_Float16)x2.z; a[11] = (_Float16)x2.w;
    a[12]= (_Float16)x3.x; a[13] = (_Float16)x3.y; a[14] = (_Float16)x3.z; a[15] = (_Float16)x3.w;
    return a;
}

// A (16x32 MxK, 16-bit): lane L<16 : m=L,    halves 0-7 = K0..7,  8-15 = K16..23
//                        lane L>=16: m=L-16, halves 0-7 = K8..15, 8-15 = K24..31
// pA = A + m*ldk + hk   (hk = (lane>>4)*8, folded into the pointer)
__device__ __forceinline__ v16h load_A_frag(const float* __restrict__ pA, int kc) {
    return cvt16(*(const float4*)(pA + kc),
                 *(const float4*)(pA + kc + 4),
                 *(const float4*)(pA + kc + 16),
                 *(const float4*)(pA + kc + 20));
}

// B (32x16 KxN, 16-bit): lane L: col = L&15, halves 0-7 = K (L>=16 ? 8:0)+j, 8-15 = K 16+(L>=16 ? 8:0)+j-8
// pB = W + r*ldk + ((lane>>4)*8): same K order as A.
__device__ __forceinline__ v16h load_B_frag(const float* __restrict__ pB, int kc) {
    return cvt16(*(const float4*)(pB + kc),
                 *(const float4*)(pB + kc + 4),
                 *(const float4*)(pB + kc + 16),
                 *(const float4*)(pB + kc + 20));
}

// out[n, r] = sum_k A1[n,k] W1[r,k] (+ sum_k A2[n,k] W2[r,k]) + b1[r] (+ b2[r])
// One wave per 16-row r-tile, TWO accumulators (n=0..15, n=16..31) so each
// streamed weight fragment is loaded once and feeds two WMMAs.
template <int K1, int K2, bool DUAL, bool BIAS2>
__global__ __launch_bounds__(256) void k_gemm(const float* __restrict__ A1,
                                              const float* __restrict__ W1,
                                              const float* __restrict__ A2,
                                              const float* __restrict__ W2,
                                              const float* __restrict__ b1,
                                              const float* __restrict__ b2,
                                              float* __restrict__ out, int R) {
    int w    = threadIdx.x >> 5;
    int lane = threadIdx.x & 31;
    int r0   = (blockIdx.x * 8 + w) * 16;
    int rr   = r0 + (lane & 15);
    if (rr > R - 1) rr = R - 1;                 // clamp loads on ragged last tile
    int hk   = (lane >> 4) * 8;
    int bcol = (lane >> 4) * 8;

    v8f acc0 = {}, acc1 = {};
    {
        const float* pA0 = A1 + (size_t)(lane & 15) * K1 + hk;        // n 0..15
        const float* pA1 = pA0 + (size_t)16 * K1;                     // n 16..31
        const float* pB  = W1 + (size_t)rr * K1 + bcol;
        for (int kc = 0; kc < K1; kc += 32) {
            v16h b  = load_B_frag(pB, kc);
            v16h a0 = load_A_frag(pA0, kc);
            v16h a1 = load_A_frag(pA1, kc);
            acc0 = __builtin_amdgcn_wmma_f32_16x16x32_f16(false, a0, false, b, (short)0, acc0, false, false);
            acc1 = __builtin_amdgcn_wmma_f32_16x16x32_f16(false, a1, false, b, (short)0, acc1, false, false);
        }
    }
    if constexpr (DUAL) {
        const float* pA0 = A2 + (size_t)(lane & 15) * K2 + hk;
        const float* pA1 = pA0 + (size_t)16 * K2;
        const float* pB  = W2 + (size_t)rr * K2 + bcol;
        for (int kc = 0; kc < K2; kc += 32) {
            v16h b  = load_B_frag(pB, kc);
            v16h a0 = load_A_frag(pA0, kc);
            v16h a1 = load_A_frag(pA1, kc);
            acc0 = __builtin_amdgcn_wmma_f32_16x16x32_f16(false, a0, false, b, (short)0, acc0, false, false);
            acc1 = __builtin_amdgcn_wmma_f32_16x16x32_f16(false, a1, false, b, (short)0, acc1, false, false);
        }
    }
    // D layout: lane L: col = L&15, row(n) = g + 8*(L>>4) for VGPR g
    int col = lane & 15;
    int r = r0 + col;
    if (r < R) {
        float bb = b1[r];
        if constexpr (BIAS2) bb += b2[r];
        #pragma unroll
        for (int g = 0; g < 8; ++g) {
            int n = g + ((lane >> 4) << 3);
            out[(size_t)n * R + r]        = acc0[g] + bb;
            out[(size_t)(n + 16) * R + r] = acc1[g] + bb;
        }
    }
}

// ---------------------------------------------------------------------------
// LSTM cell (PyTorch gate order i,f,g,o), writes h1/c1 straight into d_out
__global__ void k_lstm(const float* __restrict__ gates, const float* __restrict__ c0,
                       float* __restrict__ h1, float* __restrict__ c1) {
    int i = blockIdx.x * blockDim.x + threadIdx.x;   // 32*512
    int n = i >> 9, h = i & (H_ - 1);
    const float* g = gates + (size_t)n * 4 * H_;
    float gi = sigmoidf_(g[h]);
    float gf = sigmoidf_(g[H_ + h]);
    float gg = tanhf(g[2 * H_ + h]);
    float go = sigmoidf_(g[3 * H_ + h]);
    float c  = gf * c0[i] + gi * gg;
    c1[i] = c;
    h1[i] = go * tanhf(c);
}

// ---------------------------------------------------------------------------
// logp[n, v] = logits[n, v] - max_v - log(sum exp(. - max))
__global__ void k_logsoftmax(const float* __restrict__ logits, float* __restrict__ out) {
    int n = blockIdx.x;
    const float* row = logits + (size_t)n * V_;
    __shared__ float red[256];
    float m = -1e30f;
    for (int v = threadIdx.x; v < V_; v += 256) m = fmaxf(m, row[v]);
    red[threadIdx.x] = m; __syncthreads();
    for (int s = 128; s > 0; s >>= 1) {
        if (threadIdx.x < s) red[threadIdx.x] = fmaxf(red[threadIdx.x], red[threadIdx.x + s]);
        __syncthreads();
    }
    m = red[0]; __syncthreads();
    float sum = 0.0f;
    for (int v = threadIdx.x; v < V_; v += 256) sum += expf(row[v] - m);
    red[threadIdx.x] = sum; __syncthreads();
    for (int s = 128; s > 0; s >>= 1) {
        if (threadIdx.x < s) red[threadIdx.x] += red[threadIdx.x + s];
        __syncthreads();
    }
    float lse = m + logf(red[0]);
    for (int v = threadIdx.x; v < V_; v += 256) out[(size_t)n * V_ + v] = row[v] - lse;
}

// ---------------------------------------------------------------------------
extern "C" void kernel_launch(void* const* d_in, const int* in_sizes, int n_in,
                              void* d_out, int out_size, void* d_ws, size_t ws_size,
                              hipStream_t stream) {
    const int*   ids  = (const int*)d_in[0];
    const float* h0   = (const float*)d_in[1];
    const float* c0   = (const float*)d_in[2];
    const float* enc  = (const float*)d_in[3];
    const float* wf   = (const float*)d_in[4];
    const float* mask = (const float*)d_in[5];
    const float* emb  = (const float*)d_in[6];
    const float* rW   = (const float*)d_in[7];
    const float* rb   = (const float*)d_in[8];
    const float* uW   = (const float*)d_in[9];
    const float* ub   = (const float*)d_in[10];
    const float* Wih  = (const float*)d_in[11];
    const float* Whh  = (const float*)d_in[12];
    const float* bih  = (const float*)d_in[13];
    const float* bhh  = (const float*)d_in[14];
    const float* outW = (const float*)d_in[15];
    const float* outb = (const float*)d_in[16];

    float* out = (float*)d_out;
    float* ws  = (float*)d_ws;

    // workspace layout (floats): ~7.3 MB total
    float* rt     = ws;                       // N*D
    float* sc     = rt + N_ * D_;             // N*T (scores -> attention weights)
    float* part   = sc + N_ * T_;             // N*8*C (context partials)
    float* ctx    = part + N_ * 8 * C_;       // N*C
    float* x      = ctx + N_ * C_;            // N*(E+C)
    float* gates  = x + N_ * (E_ + C_);       // N*4H
    float* logits = gates + N_ * 4 * H_;      // N*V

    float* logp = out;                        // [N, 1, V]
    float* h1   = out + (size_t)N_ * V_;      // [1, N, H]
    float* c1   = h1 + N_ * H_;               // [1, N, H]

    k_rt<<<N_, 256, 0, stream>>>(h0, rW, rb, rt);
    k_scores<<<(N_ * T_) / 8, 256, 0, stream>>>(wf, rt, uW, ub, mask, sc);
    k_softmax<<<N_, 256, 0, stream>>>(sc);
    k_context<<<N_ * 8, 256, 0, stream>>>(sc, enc, part);
    k_ctx_reduce<<<(N_ * C_) / 256, 256, 0, stream>>>(part, ctx);
    k_build_x<<<N_, 256, 0, stream>>>(ids, emb, ctx, x);
    // gates: 2048/16 = 128 r-tiles, 8 waves/block -> 16 blocks
    k_gemm<E_ + C_, H_, true, true><<<16, 256, 0, stream>>>(x, Wih, h0, Whh,
                                                            bih, bhh, gates, 4 * H_);
    k_lstm<<<(N_ * H_) / 256, 256, 0, stream>>>(gates, c0, h1, c1);
    // logits: ceil(50257/16)=3142 r-tiles, 8/block -> 393 blocks
    int rtiles = (V_ + 15) / 16;
    k_gemm<H_, 0, false, false><<<(rtiles + 7) / 8, 256, 0, stream>>>(h1, outW, nullptr, nullptr,
                                                                      outb, nullptr, logits, V_);
    k_logsoftmax<<<N_, 256, 0, stream>>>(logits, logp);
}
